// imgNet_47974784696694
// MI455X (gfx1250) — matrix operand order repaired and checked by reading, not compiled, not run on hardware
//
#include <hip/hip_runtime.h>
#include <hip/hip_bf16.h>

// ---------------------------------------------------------------------------
// MI455X (gfx1250): bf16 WMMA GEMMs with f32 accumulate.
//   - All operands pre-converted to bf16 ONCE (conversion out of hot loop).
//   - GEMM hot loop (preferred path): GLOBAL_LOAD_ASYNC_TO_LDS_B128 DMA of
//     the next tile into the ping-pong LDS buffer overlapping 8x
//     v_wmma_f32_16x16x32_bf16 on the current tile; s_wait_asynccnt + barrier
//     hand off buffers.  Falls back to global_load_b128 -> ds_store_b128
//     register staging when the async builtin is unavailable.
//   - RNN: 59/64 steps use fused Wsum = W_ih + W_hh (x == h), 5 boundary
//     steps accumulate two GEMMs (static schedule from the reference).
// ---------------------------------------------------------------------------

typedef __attribute__((ext_vector_type(16))) __bf16 v16bf;
typedef __attribute__((ext_vector_type(8)))  float  v8f;

#define BM 128
#define BN 128
#define BKK 32
#define LSTR 40           // padded LDS row stride (bf16 elems); 80B, 16B-aligned
#define HIDN 1024
#define BATCH 2048
#define FDIM 4096

typedef unsigned short bf_t;   // bf16 storage

// ---- optional gfx1250 async global->LDS path (probe via __has_builtin) ----
#if defined(__has_builtin)
#  if __has_builtin(__builtin_amdgcn_global_load_async_to_lds_b128)
#    define USE_ASYNC_LDS 1
#  endif
#endif
#ifndef USE_ASYNC_LDS
#  define USE_ASYNC_LDS 0
#endif

#if USE_ASYNC_LDS
// builtin signature (from clang diagnostic): (int4 AS1*, int4 AS3*, Imm, Imm)
typedef int v4i_ __attribute__((vector_size(16)));
#define ASYNC_CP_B128(GSRC, LDST)                                             \
  __builtin_amdgcn_global_load_async_to_lds_b128(                             \
      (__attribute__((address_space(1))) v4i_*)(GSRC),                        \
      (__attribute__((address_space(3))) v4i_*)(LDST), 0, 0)
#if __has_builtin(__builtin_amdgcn_s_wait_asynccnt)
#define ASYNC_WAIT() __builtin_amdgcn_s_wait_asynccnt(0)
#else
#define ASYNC_WAIT() asm volatile("s_wait_asynccnt 0" ::: "memory")
#endif
#endif

__device__ __forceinline__ unsigned f2bf(float f) {
  // round-to-nearest-even f32 -> bf16 (bit trick; only used OUTSIDE hot loops)
  unsigned u = __builtin_bit_cast(unsigned, f);
  return (u + 0x7FFFu + ((u >> 16) & 1u)) >> 16;
}
__device__ __forceinline__ unsigned pack2bf(float a, float b) {
  return (f2bf(a) & 0xFFFFu) | (f2bf(b) << 16);
}

union Frag { uint4 q[2]; v16bf v; };

// ACT: 0 = none, 1 = relu, 2 = tanh.  DUAL: C = act(A0*B0^T + A1*B1^T + b).
// A: [M,K] bf16 row-major.  B: [N,K] bf16 row-major (i.e. W[out,in]).
template<int ACT, bool DUAL, bool WRITE_F32>
__global__ __launch_bounds__(256) void gemm_bf16_wmma(
    const bf_t* __restrict__ A0, const bf_t* __restrict__ B0,
    const bf_t* __restrict__ A1, const bf_t* __restrict__ B1,
    const float* __restrict__ bias1, const float* __restrict__ bias2,
    float* __restrict__ Cf, bf_t* __restrict__ Cbf, int N, int K)
{
  __shared__ __align__(16) bf_t As[2][BM * LSTR];
  __shared__ __align__(16) bf_t Bs[2][BN * LSTR];

  const int tid  = threadIdx.x;
  const int lane = tid & 31;
  const int wave = tid >> 5;
  const int wy = wave >> 1;        // 4 waves over M -> 32-row strips
  const int wx = wave & 1;         // 2 waves over N -> 64-col strips
  const int lm = lane & 15;
  const int lh = (lane >> 4) & 1;  // lane half selects K sub-block

  const int m0 = blockIdx.y * BM;
  const int n0 = blockIdx.x * BN;

  // ---- loop-invariant staging offsets (derived once from tid) ----
  const int srow = tid >> 2;                // 0..63 (first of two rows)
  const int scol = (tid & 3) * 8;           // bf16 col within 32-wide tile
  const size_t gA  = (size_t)(m0 + srow) * K + scol;   // + k0 per tile
  const size_t gB  = (size_t)(n0 + srow) * K + scol;
  const size_t gRowStep = (size_t)64 * K;              // rows srow+64
  const int    lOff  = srow * LSTR + scol;             // LDS slot
  const int    lOff2 = lOff + 64 * LSTR;

  // ---- loop-invariant fragment LDS offsets ----
  const int faOff0 = (wy * 32 +  0 + lm) * LSTR + (lh ? 8 : 0);
  const int faOff1 = (wy * 32 + 16 + lm) * LSTR + (lh ? 8 : 0);
  const int fbBase = (wx * 64 + lm) * LSTR + (lh ? 8 : 0);

  v8f acc[2][4];
  const v8f vzero = {0.f, 0.f, 0.f, 0.f, 0.f, 0.f, 0.f, 0.f};
#pragma unroll
  for (int i = 0; i < 2; ++i)
#pragma unroll
    for (int j = 0; j < 4; ++j) acc[i][j] = vzero;

  const int tilesPerPass = K / BKK;
  const int nIter = (DUAL ? 2 : 1) * tilesPerPass;

#define TILE_SRC(IT, AP, BP, K0)                                              \
  const bf_t* AP = A0;                                                        \
  const bf_t* BP = B0;                                                        \
  int K0 = (IT) * BKK;                                                        \
  if (DUAL && (IT) >= tilesPerPass) {                                         \
    AP = A1; BP = B1; K0 = ((IT) - tilesPerPass) * BKK;                       \
  }

#define COMPUTE_TILE(BUF)                                                     \
  do {                                                                        \
    Frag fa0, fa1;                                                            \
    {                                                                         \
      const bf_t* p0 = &As[BUF][faOff0];                                      \
      const bf_t* p1 = &As[BUF][faOff1];                                      \
      fa0.q[0] = *(const uint4*)p0;                                           \
      fa0.q[1] = *(const uint4*)(p0 + 16);                                    \
      fa1.q[0] = *(const uint4*)p1;                                           \
      fa1.q[1] = *(const uint4*)(p1 + 16);                                    \
    }                                                                         \
    Frag fb[4];                                                               \
    _Pragma("unroll")                                                         \
    for (int j = 0; j < 4; ++j) {                                             \
      const bf_t* p = &Bs[BUF][fbBase + j * 16 * LSTR];                       \
      fb[j].q[0] = *(const uint4*)p;                                          \
      fb[j].q[1] = *(const uint4*)(p + 16);                                   \
    }                                                                         \
    _Pragma("unroll")                                                         \
    for (int j = 0; j < 4; ++j)                                               \
      acc[0][j] = __builtin_amdgcn_wmma_f32_16x16x32_bf16(                    \
          false, fa0.v, false, fb[j].v, (short)0, acc[0][j], false, false);   \
    _Pragma("unroll")                                                         \
    for (int j = 0; j < 4; ++j)                                               \
      acc[1][j] = __builtin_amdgcn_wmma_f32_16x16x32_bf16(                    \
          false, fa1.v, false, fb[j].v, (short)0, acc[1][j], false, false);   \
  } while (0)

#if USE_ASYNC_LDS
  // -------- async DMA path: global -> LDS without touching VGPRs ----------
#define ISSUE_TILE(IT, BUF)                                                   \
  do {                                                                        \
    TILE_SRC(IT, Ap_, Bp_, k0_)                                               \
    ASYNC_CP_B128(Ap_ + gA + k0_,            &As[BUF][lOff]);                 \
    ASYNC_CP_B128(Ap_ + gA + gRowStep + k0_, &As[BUF][lOff2]);                \
    ASYNC_CP_B128(Bp_ + gB + k0_,            &Bs[BUF][lOff]);                 \
    ASYNC_CP_B128(Bp_ + gB + gRowStep + k0_, &Bs[BUF][lOff2]);                \
  } while (0)

  ISSUE_TILE(0, 0);
  ASYNC_WAIT();
  __syncthreads();
  int buf = 0;

  for (int it = 0; it < nIter; ++it) {
    const bool more = (it + 1) < nIter;
    if (more) ISSUE_TILE(it + 1, buf ^ 1);  // DMA overlaps the WMMAs below
    COMPUTE_TILE(buf);
    if (more) ASYNC_WAIT();                 // own tile-portion landed in LDS
    __syncthreads();                        // all waves' portions visible
    buf ^= 1;
  }
#undef ISSUE_TILE

#else
  // -------- fallback: register staging (named scalars, no scratch) --------
  uint4 ta0, ta1, tb0, tb1;

#define LOAD_TILE(IT)                                                         \
  do {                                                                        \
    TILE_SRC(IT, Ap_, Bp_, k0_)                                               \
    ta0 = *(const uint4*)(Ap_ + gA + k0_);                                    \
    ta1 = *(const uint4*)(Ap_ + gA + gRowStep + k0_);                         \
    tb0 = *(const uint4*)(Bp_ + gB + k0_);                                    \
    tb1 = *(const uint4*)(Bp_ + gB + gRowStep + k0_);                         \
  } while (0)

#define STORE_TILE(BUF)                                                       \
  do {                                                                        \
    *(uint4*)&As[BUF][lOff]  = ta0;                                           \
    *(uint4*)&As[BUF][lOff2] = ta1;                                           \
    *(uint4*)&Bs[BUF][lOff]  = tb0;                                           \
    *(uint4*)&Bs[BUF][lOff2] = tb1;                                           \
  } while (0)

  LOAD_TILE(0);
  STORE_TILE(0);
  int buf = 0;

  for (int it = 0; it < nIter; ++it) {
    const bool more = (it + 1) < nIter;
    if (more) LOAD_TILE(it + 1);   // global loads in flight across the barrier
    __syncthreads();               // previous iteration's stores visible
    COMPUTE_TILE(buf);
    if (more) STORE_TILE(buf ^ 1); // fill other buffer
    buf ^= 1;
  }
#undef LOAD_TILE
#undef STORE_TILE
#endif
#undef COMPUTE_TILE
#undef TILE_SRC

  // ---- epilogue: bias + activation; bf16 always, f32 if WRITE_F32 ----
#pragma unroll
  for (int i = 0; i < 2; ++i) {
#pragma unroll
    for (int j = 0; j < 4; ++j) {
      const int n = n0 + wx * 64 + j * 16 + lm;
      float b = 0.f;
      if (bias1) b += bias1[n];
      if (bias2) b += bias2[n];
#pragma unroll
      for (int r = 0; r < 8; ++r) {
        const int m = m0 + wy * 32 + i * 16 + r + (lh ? 8 : 0);
        float v = acc[i][j][r] + b;
        if (ACT == 1) v = fmaxf(v, 0.f);
        if (ACT == 2) v = tanhf(v);
        Cbf[(size_t)m * N + n] = (bf_t)f2bf(v);
        if (WRITE_F32) Cf[(size_t)m * N + n] = v;
      }
    }
  }
}

// f32 -> bf16 bulk convert (one float4 -> 4 bf16 per thread)
__global__ void cvt_bf16(const float* __restrict__ src, bf_t* __restrict__ dst,
                         int n4) {
  int i = blockIdx.x * blockDim.x + threadIdx.x;
  if (i < n4) {
    float4 v = ((const float4*)src)[i];
    uint2 p; p.x = pack2bf(v.x, v.y); p.y = pack2bf(v.z, v.w);
    ((uint2*)dst)[i] = p;
  }
}

// Wsum = bf16(W_ih + W_hh): fused add+convert
__global__ void sum_cvt_bf16(const float* __restrict__ a,
                             const float* __restrict__ b,
                             bf_t* __restrict__ dst, int n4) {
  int i = blockIdx.x * blockDim.x + threadIdx.x;
  if (i < n4) {
    float4 x = ((const float4*)a)[i];
    float4 y = ((const float4*)b)[i];
    uint2 p;
    p.x = pack2bf(x.x + y.x, x.y + y.y);
    p.y = pack2bf(x.z + y.z, x.w + y.w);
    ((uint2*)dst)[i] = p;
  }
}

// probs[row] = sigmoid(hx[row,:] . W_act + b_act); one wave32 per row
__global__ __launch_bounds__(256) void head_sigmoid(
    const float* __restrict__ hx, const float* __restrict__ Wact,
    const float* __restrict__ bact, float* __restrict__ out)
{
  const int row  = blockIdx.x * 8 + (threadIdx.x >> 5);
  const int lane = threadIdx.x & 31;
  const float* h = hx + (size_t)row * HIDN;
  float s = 0.f;
#pragma unroll 4
  for (int k = lane; k < HIDN; k += 32) s += h[k] * Wact[k];
#pragma unroll
  for (int off = 16; off > 0; off >>= 1) s += __shfl_down(s, off, 32);
  if (lane == 0) out[row] = 1.f / (1.f + expf(-(s + bact[0])));
}

__global__ void copy_f32(const float* __restrict__ src, float* __restrict__ dst,
                         int n) {
  int i = blockIdx.x * blockDim.x + threadIdx.x;
  if (i < n) dst[i] = src[i];
}

extern "C" void kernel_launch(void* const* d_in, const int* in_sizes, int n_in,
                              void* d_out, int out_size, void* d_ws, size_t ws_size,
                              hipStream_t stream) {
  const float* x_low  = (const float*)d_in[0];
  const float* x_mid  = (const float*)d_in[1];
  const float* x_hig  = (const float*)d_in[2];
  const float* low_W1 = (const float*)d_in[3];
  const float* low_b1 = (const float*)d_in[4];
  const float* low_W2 = (const float*)d_in[5];
  const float* low_b2 = (const float*)d_in[6];
  const float* mid_W1 = (const float*)d_in[7];
  const float* mid_b1 = (const float*)d_in[8];
  const float* mid_W2 = (const float*)d_in[9];
  const float* mid_b2 = (const float*)d_in[10];
  const float* hig_W1 = (const float*)d_in[11];
  const float* hig_b1 = (const float*)d_in[12];
  const float* hig_W2 = (const float*)d_in[13];
  const float* hig_b2 = (const float*)d_in[14];
  const float* W_ih   = (const float*)d_in[15];
  const float* W_hh   = (const float*)d_in[16];
  const float* b_ih   = (const float*)d_in[17];
  const float* b_hh   = (const float*)d_in[18];
  const float* W_act  = (const float*)d_in[19];
  const float* b_act  = (const float*)d_in[20];
  float* out = (float*)d_out;

  // ---- workspace layout ----
  const size_t XSZ = (size_t)BATCH * FDIM;   // 8M elems
  const size_t HSZ = (size_t)BATCH * HIDN;   // 2M elems
  const size_t W1S = (size_t)HIDN * FDIM;    // 4M elems
  const size_t W2S = (size_t)HIDN * HIDN;    // 1M elems

  bf_t* p = (bf_t*)d_ws;
  bf_t* xbf[3]   = {p, p + XSZ, p + 2 * XSZ};          p += 3 * XSZ;
  bf_t* W1bf[3]  = {p, p + W1S, p + 2 * W1S};          p += 3 * W1S;
  bf_t* W2bf[3]  = {p, p + W2S, p + 2 * W2S};          p += 3 * W2S;
  bf_t* Wihbf    = p;                                  p += W2S;
  bf_t* Whhbf    = p;                                  p += W2S;
  bf_t* Wsumbf   = p;                                  p += W2S;
  bf_t* h1bf     = p;                                  p += HSZ;
  bf_t* actbf[3] = {p, p + HSZ, p + 2 * HSZ};          p += 3 * HSZ;  // xl,xm,xh
  bf_t* hxbf[2]  = {p, p + HSZ};                       p += 2 * HSZ;
  float* hxf[2];
  hxf[0] = (float*)p;
  hxf[1] = hxf[0] + HSZ;

  const dim3 blk(256);
  const dim3 grd(HIDN / BN, BATCH / BM);   // 8 x 16 blocks

  // ---- one-time conversions (off the hot path) ----
  const float* xs[3]  = {x_low, x_mid, x_hig};
  const float* W1s[3] = {low_W1, mid_W1, hig_W1};
  const float* W2s[3] = {low_W2, mid_W2, hig_W2};
  for (int b = 0; b < 3; ++b) {
    cvt_bf16<<<(int)(XSZ / 4 + 255) / 256, 256, 0, stream>>>(xs[b], xbf[b], (int)(XSZ / 4));
    cvt_bf16<<<(int)(W1S / 4 + 255) / 256, 256, 0, stream>>>(W1s[b], W1bf[b], (int)(W1S / 4));
    cvt_bf16<<<(int)(W2S / 4 + 255) / 256, 256, 0, stream>>>(W2s[b], W2bf[b], (int)(W2S / 4));
  }
  cvt_bf16<<<(int)(W2S / 4 + 255) / 256, 256, 0, stream>>>(W_ih, Wihbf, (int)(W2S / 4));
  cvt_bf16<<<(int)(W2S / 4 + 255) / 256, 256, 0, stream>>>(W_hh, Whhbf, (int)(W2S / 4));
  sum_cvt_bf16<<<(int)(W2S / 4 + 255) / 256, 256, 0, stream>>>(W_ih, W_hh, Wsumbf, (int)(W2S / 4));

  // ---- three MLP branches (GEMM+ReLU, GEMM+ReLU), bf16 outputs only ----
  const float* b1s[3] = {low_b1, mid_b1, hig_b1};
  const float* b2s[3] = {low_b2, mid_b2, hig_b2};
  for (int b = 0; b < 3; ++b) {
    gemm_bf16_wmma<1, false, false><<<grd, blk, 0, stream>>>(
        xbf[b], W1bf[b], nullptr, nullptr, b1s[b], nullptr,
        nullptr, h1bf, HIDN, FDIM);
    gemm_bf16_wmma<1, false, false><<<grd, blk, 0, stream>>>(
        h1bf, W2bf[b], nullptr, nullptr, b2s[b], nullptr,
        nullptr, actbf[b], HIDN, HIDN);
  }

  // ---- 64-step RNN (static schedule; replicate reference tip logic) ----
  const bf_t* hx  = actbf[2];   // hx starts as xh (bf16 GEMM input)
  const bf_t* ipt = actbf[2];
  int cur = 0, tip = 1;
  for (int step = 0; step < 64; ++step) {
    bool boundary = false;
    if (step == 21)         { ipt = actbf[1]; boundary = true; tip = 1; }
    else if (step == 42)    { ipt = actbf[0]; boundary = true; tip = 1; }
    else if (tip % 16 == 0) { boundary = true; tip = 1; }
    else                    { tip += 1; }

    bf_t*  hxo_bf = hxbf[cur];
    float* hxo_f  = hxf[cur];
    cur ^= 1;
    if (boundary) {
      // hx' = tanh(ipt @ W_ih^T + hx @ W_hh^T + b_ih + b_hh)
      gemm_bf16_wmma<2, true, true><<<grd, blk, 0, stream>>>(
          ipt, Wihbf, hx, Whhbf, b_ih, b_hh, hxo_f, hxo_bf, HIDN, HIDN);
    } else {
      // x == h  =>  hx' = tanh(hx @ (W_ih+W_hh)^T + b)   (half the FLOPs)
      gemm_bf16_wmma<2, false, true><<<grd, blk, 0, stream>>>(
          hx, Wsumbf, nullptr, nullptr, b_ih, b_hh, hxo_f, hxo_bf, HIDN, HIDN);
    }
    head_sigmoid<<<BATCH / 8, 256, 0, stream>>>(hxo_f, W_act, b_act,
                                                out + (size_t)step * BATCH);
    hx = hxo_bf;
  }

  // final hx (f32) appended after the 64*2048 probs
  const float* hx_final = hxf[cur ^ 1];
  copy_f32<<<((int)HSZ + 255) / 256, 256, 0, stream>>>(hx_final,
                                                       out + 64 * BATCH,
                                                       (int)HSZ);
}
